// ACA_GCN_25580825215621
// MI455X (gfx1250) — matrix operand order repaired and checked by reading, not compiled, not run on hardware
//
#include <hip/hip_runtime.h>
#include <hip/hip_bf16.h>

typedef __bf16 bf16_t;
typedef __attribute__((ext_vector_type(16))) __bf16 v16bf;
typedef __attribute__((ext_vector_type(8)))  float  v8f;
typedef __attribute__((ext_vector_type(4)))  unsigned int u32x4;
typedef __attribute__((ext_vector_type(8)))  unsigned int u32x8;

#define D_IN 128
#define HGC 64

// ---------------------------------------------------------------- weight swizzle
// Convert f32 K x N weight to bf16 in WMMA-B-fragment-major order:
// dst[(((k>>5)*N + n)*2 + ((k>>3)&1))*16 + ((k&7)|((k>>1)&8))] = src[k*N + n]
// so each lane's 16 B elements (K = 32s + 8h + j, j<8; 32s + 16 + 8h + j-8, j>=8; fixed n) are contiguous (32B).
__global__ __launch_bounds__(256) void cvt_swz_kernel(
    const float* __restrict__ src, bf16_t* __restrict__ dst, int K, int ncols) {
  int i = blockIdx.x * 256 + threadIdx.x;
  if (i >= K * ncols) return;
  int k = i / ncols, n = i - k * ncols;
  int s = k >> 5, h = (k >> 3) & 1, j = (k & 7) | ((k >> 1) & 8);
  dst[((size_t)(s * ncols + n) * 2 + h) * 16 + j] = (bf16_t)src[i];
}

// ---------------------------------------------------------------- TDM tile load
// Issue a Tensor-Data-Mover DMA of a 16-row x 256-float tile (row stride 256
// floats) from global memory into LDS at lds_off. 2D descriptor: groups 0+1.
__device__ __forceinline__ void tdm_load_tile(const float* gsrc, unsigned lds_off) {
  unsigned long long ga = (unsigned long long)(size_t)gsrc;
  u32x4 g0;
  g0[0] = 1u;                                   // count=1, user descriptor
  g0[1] = lds_off;                              // lds_addr (bytes)
  g0[2] = (unsigned)ga;                         // global_addr[31:0]
  g0[3] = ((unsigned)(ga >> 32) & 0x01FFFFFFu)  // global_addr[56:32]
          | 0x80000000u;                        // type=2 ("image") in bits 127:126
  u32x8 g1;
  g1[0] = 0x00020000u;      // workgroup_mask=0, data_size=2 (4B), no pad/iterate
  g1[1] = 256u << 16;       // tensor_dim0 = 256 elements (bits 79:48, low half)
  g1[2] = 16u << 16;        // tensor_dim0 hi=0 | tensor_dim1 = 16 (bits 111:80)
  g1[3] = 256u << 16;       // tensor_dim1 hi=0 | tile_dim0 = 256 (bits 127:112)
  g1[4] = 16u;              // tile_dim1 = 16, tile_dim2 = 0
  g1[5] = 256u;             // tensor_dim0_stride = 256 (low 32 of 48)
  g1[6] = 0u;               // stride hi, tensor_dim1_stride lo
  g1[7] = 0u;
  asm volatile("tensor_load_to_lds %0, %1" :: "s"(g0), "s"(g1) : "memory");
}

// ---------------------------------------------------------------- PAE: edge MLP + cosine
// 256 threads = 8 waves; 64 edges per block (4 tiles of 16).
// Wave w owns output columns [16w, 16w+16). Both MLP halves share B fragments.
// Edge tiles are double-buffered in LDS via TDM async DMA (wave 0 issues,
// TENSORcnt + barrier publishes). Weights live in LDS fragment-major, padded.
#define WSLOT 40  // bf16 elements per (s,col): 2 halves * 16 + 8 pad
__global__ __launch_bounds__(256) void pae_kernel(
    const float* __restrict__ xin,      // E x 256
    const bf16_t* __restrict__ w1g,     // swizzled 128x128 bf16
    const float* __restrict__ b1,
    const bf16_t* __restrict__ w2g,     // swizzled 128x128 bf16
    const float* __restrict__ b2,
    float* __restrict__ ew, int E) {
  __shared__ __align__(32) bf16_t ws1[4 * 128 * WSLOT];
  __shared__ __align__(32) bf16_t ws2[4 * 128 * WSLOT];
  __shared__ __align__(16) bf16_t xs[2][16 * 136];
  __shared__ __align__(16) bf16_t hs[2][16 * 136];
  __shared__ __align__(16) float xstage[2 * 16 * 256];   // TDM double buffer
  __shared__ float dots[16][4];

  const int tid  = threadIdx.x;
  const int wv   = tid >> 5;
  const int lane = tid & 31;

  // stage swizzled weights: straight b128 fragment copies
  for (int f = tid; f < 1024; f += 256) {
    const uint4* s1 = (const uint4*)(w1g + (size_t)f * 16);
    const uint4* s2 = (const uint4*)(w2g + (size_t)f * 16);
    uint4* d1 = (uint4*)&ws1[(f >> 1) * WSLOT + (f & 1) * 16];
    uint4* d2 = (uint4*)&ws2[(f >> 1) * WSLOT + (f & 1) * 16];
    d1[0] = s1[0]; d1[1] = s1[1];
    d2[0] = s2[0]; d2[1] = s2[1];
  }

  // kick off TDM for tile 0 while weights stage
  const int e00 = blockIdx.x * 64;
  if (wv == 0 && e00 + 16 <= E)
    tdm_load_tile(xin + (size_t)e00 * 256, (unsigned)(size_t)&xstage[0]);
  __syncthreads();

  const int nloc = lane & 15;
  const int col  = nloc + 16 * wv;
  const int half = (lane >> 4) & 1;
  const int kbA  = half * 8;          // A frag: K base per half-wave
  const int moff = half * 8;          // C frag: row base per half-wave
  const float bias1 = b1[col];
  const float bias2 = b2[col];

  for (int tile = 0; tile < 4; ++tile) {
    const int e0 = (blockIdx.x * 4 + tile) * 16;
    const bool full = (e0 + 16 <= E);

    // wave 0: prefetch next tile via TDM, then wait for current tile's DMA
    if (wv == 0) {
      bool issued_next = false;
      const int nt = tile + 1;
      if (nt < 4) {
        const int e0n = (blockIdx.x * 4 + nt) * 16;
        if (e0n + 16 <= E) {
          tdm_load_tile(xin + (size_t)e0n * 256,
                        (unsigned)(size_t)&xstage[(nt & 1) * 4096]);
          issued_next = true;
        }
      }
      if (issued_next) __builtin_amdgcn_s_wait_tensorcnt(1);
      else             __builtin_amdgcn_s_wait_tensorcnt(0);
    }
    __syncthreads();   // publish TDM-written LDS tile to all waves

    // convert current tile f32 -> bf16 A-tile layout
    const float* xsrc = &xstage[(tile & 1) * 4096];
    for (int i = tid; i < 16 * 256; i += 256) {
      int m = i >> 8, c = i & 255;
      float v = full ? xsrc[i]
                     : ((e0 + m < E) ? xin[(size_t)(e0 + m) * 256 + c] : 0.0f);
      xs[c >> 7][m * 136 + (c & 127)] = (bf16_t)v;
    }
    if (tid < 16) { dots[tid][0] = 0.f; dots[tid][1] = 0.f; dots[tid][2] = 0.f; }
    __syncthreads();

    // ---- layer 1: h_pre = relu(x @ w1 + b1), halves share B fragments
    v8f acc1 = {}, acc2 = {};
#pragma unroll
    for (int s = 0; s < 4; ++s) {
      const v16bf bfrag = *(const v16bf*)&ws1[(s * 128 + col) * WSLOT + half * 16];
      const int ab = nloc * 136 + s * 32 + kbA;
      v16bf a1, a2;
#pragma unroll
      for (int j = 0; j < 8; ++j) {
        a1[j] = xs[0][ab + j];      a1[j + 8] = xs[0][ab + 16 + j];
        a2[j] = xs[1][ab + j];      a2[j + 8] = xs[1][ab + 16 + j];
      }
      acc1 = __builtin_amdgcn_wmma_f32_16x16x32_bf16(false, a1, false, bfrag, (short)0, acc1, false, false);
      acc2 = __builtin_amdgcn_wmma_f32_16x16x32_bf16(false, a2, false, bfrag, (short)0, acc2, false, false);
    }
#pragma unroll
    for (int r = 0; r < 8; ++r) {
      int m = moff + r;
      hs[0][m * 136 + col] = (bf16_t)fmaxf(acc1[r] + bias1, 0.0f);
      hs[1][m * 136 + col] = (bf16_t)fmaxf(acc2[r] + bias2, 0.0f);
    }
    __syncthreads();

    // ---- layer 2: h = h_pre @ w2 + b2
    v8f o1 = {}, o2 = {};
#pragma unroll
    for (int s = 0; s < 4; ++s) {
      const v16bf bfrag = *(const v16bf*)&ws2[(s * 128 + col) * WSLOT + half * 16];
      const int ab = nloc * 136 + s * 32 + kbA;
      v16bf a1, a2;
#pragma unroll
      for (int j = 0; j < 8; ++j) {
        a1[j] = hs[0][ab + j];      a1[j + 8] = hs[0][ab + 16 + j];
        a2[j] = hs[1][ab + j];      a2[j + 8] = hs[1][ab + 16 + j];
      }
      o1 = __builtin_amdgcn_wmma_f32_16x16x32_bf16(false, a1, false, bfrag, (short)0, o1, false, false);
      o2 = __builtin_amdgcn_wmma_f32_16x16x32_bf16(false, a2, false, bfrag, (short)0, o2, false, false);
    }

    // ---- cosine reduction: per edge (row) sum over all 128 columns
#pragma unroll
    for (int r = 0; r < 8; ++r) {
      float v1 = o1[r] + bias2;
      float v2 = o2[r] + bias2;
      float p12 = v1 * v2, p11 = v1 * v1, p22 = v2 * v2;
#pragma unroll
      for (int mk = 1; mk < 16; mk <<= 1) {
        p12 += __shfl_xor(p12, mk, 32);
        p11 += __shfl_xor(p11, mk, 32);
        p22 += __shfl_xor(p22, mk, 32);
      }
      if (nloc == 0) {
        atomicAdd(&dots[moff + r][0], p12);
        atomicAdd(&dots[moff + r][1], p11);
        atomicAdd(&dots[moff + r][2], p22);
      }
    }
    __syncthreads();
    if (tid < 16 && e0 + tid < E) {
      float s12 = dots[tid][0], s11 = dots[tid][1], s22 = dots[tid][2];
      float n1 = fmaxf(sqrtf(s11), 1e-8f);
      float n2 = fmaxf(sqrtf(s22), 1e-8f);
      float c = s12 / (n1 * n2);
      ew[e0 + tid] = (c + 1.0f) * 0.5f;
    }
    __syncthreads();
  }
}

// ---------------------------------------------------------------- graph normalization
__global__ __launch_bounds__(256) void deg_kernel(
    const int* __restrict__ row, const float* __restrict__ ew,
    float* __restrict__ deg, int E) {
  int e = blockIdx.x * 256 + threadIdx.x;
  if (e < E) atomicAdd(&deg[row[e]], ew[e]);
}

__global__ __launch_bounds__(256) void dinv_kernel(
    const float* __restrict__ deg, float* __restrict__ dinv, int n) {
  int i = blockIdx.x * 256 + threadIdx.x;
  if (i < n) {
    float d = deg[i];
    dinv[i] = d > 0.0f ? rsqrtf(fmaxf(d, 1e-12f)) : 0.0f;
  }
}

__global__ __launch_bounds__(256) void normc_kernel(
    const int* __restrict__ row, const int* __restrict__ col,
    const float* __restrict__ ew, const float* __restrict__ dinv,
    float* __restrict__ nrm, int E) {
  int e = blockIdx.x * 256 + threadIdx.x;
  if (e < E) nrm[e] = -dinv[row[e]] * ew[e] * dinv[col[e]];
}

// prop: out[col[e]] += nrm[e] * in[row[e]]   (float4 per thread, L2-resident gathers)
__global__ __launch_bounds__(256) void spmm_kernel(
    const int* __restrict__ row, const int* __restrict__ col,
    const float* __restrict__ nrm, const float* __restrict__ in, int lda,
    int din, float* __restrict__ out, int E) {
  int chunks = din >> 2;
  int t = blockIdx.x * 256 + threadIdx.x;
  if (t >= E * chunks) return;
  int e = t / chunks;
  int c = (t - e * chunks) * 4;
  float w = nrm[e];
  int r = row[e], d = col[e];
  const float4 v = *(const float4*)(in + (size_t)r * lda + c);
  float* o = out + (size_t)d * din + c;
  atomicAdd(o + 0, w * v.x);
  atomicAdd(o + 1, w * v.y);
  atomicAdd(o + 2, w * v.z);
  atomicAdd(o + 3, w * v.w);
}

// t2 = 2*t2 - z
__global__ __launch_bounds__(256) void axpy_kernel(
    float* __restrict__ t2, const float* __restrict__ z, int ldz, int din, int n) {
  int i = blockIdx.x * 256 + threadIdx.x;
  if (i >= n * din) return;
  int m = i / din, d = i - m * din;
  t2[i] = 2.0f * t2[i] - z[(size_t)m * ldz + d];
}

// ---------------------------------------------------------------- Cheb layer GEMM
// C(N x 64) = relu(z@W0 + t1@W1 + t2@W2), W stacked (3*din x 64) bf16 swizzled,
// written into d_out at ldc=256, column offset coloff. 128 threads = 4 waves.
__global__ __launch_bounds__(128) void cheb_gemm_kernel(
    const float* __restrict__ z, int ldz,
    const float* __restrict__ t1, const float* __restrict__ t2, int din,
    const bf16_t* __restrict__ w, float* __restrict__ out, int coloff, int n) {
  const int m0   = blockIdx.x * 16;
  const int wv   = threadIdx.x >> 5;
  const int lane = threadIdx.x & 31;
  const int nloc = lane & 15;
  const int ncol = nloc + 16 * wv;
  const int half = (lane >> 4) & 1;
  const int kbA  = half * 8;
  const int moff = half * 8;
  int mrow = m0 + nloc; if (mrow >= n) mrow = n - 1;

  v8f acc = {};
  const int nsteps = (3 * din) >> 5;
  for (int s = 0; s < nsteps; ++s) {
    const int kg  = s * 32;
    const int idx = kg / din;
    const float* Ap = (idx == 0) ? z : (idx == 1 ? t1 : t2);
    const int lda   = (idx == 0) ? ldz : din;
    const int kl    = kg - idx * din;
    const float* ap = Ap + (size_t)mrow * lda + kl + kbA;
    const float4 f0 = *(const float4*)(ap);
    const float4 f1 = *(const float4*)(ap + 4);
    const float4 f2 = *(const float4*)(ap + 16);
    const float4 f3 = *(const float4*)(ap + 20);
    v16bf a;
    a[0] = (bf16_t)f0.x; a[1] = (bf16_t)f0.y; a[2]  = (bf16_t)f0.z; a[3]  = (bf16_t)f0.w;
    a[4] = (bf16_t)f1.x; a[5] = (bf16_t)f1.y; a[6]  = (bf16_t)f1.z; a[7]  = (bf16_t)f1.w;
    a[8] = (bf16_t)f2.x; a[9] = (bf16_t)f2.y; a[10] = (bf16_t)f2.z; a[11] = (bf16_t)f2.w;
    a[12] = (bf16_t)f3.x; a[13] = (bf16_t)f3.y; a[14] = (bf16_t)f3.z; a[15] = (bf16_t)f3.w;
    const v16bf b = *(const v16bf*)(w + ((size_t)(s * 64 + ncol) * 2 + half) * 16);
    acc = __builtin_amdgcn_wmma_f32_16x16x32_bf16(false, a, false, b, (short)0, acc, false, false);
  }
#pragma unroll
  for (int r = 0; r < 8; ++r) {
    int m = m0 + moff + r;
    if (m < n) out[(size_t)m * 256 + coloff + ncol] = fmaxf(acc[r], 0.0f);
  }
}

// ---------------------------------------------------------------- classifier GEMM
// z1 = BN(relu(h0 @ W + b1)); 128 threads = 4 waves cover 64 cols; grid.y covers 256.
__global__ __launch_bounds__(128) void cls_gemm_kernel(
    const float* __restrict__ h0, const bf16_t* __restrict__ w,
    const float* __restrict__ b1, const float* __restrict__ gamma,
    const float* __restrict__ beta, const float* __restrict__ mean,
    const float* __restrict__ var, float* __restrict__ z1, int n) {
  const int m0   = blockIdx.x * 16;
  const int wv   = threadIdx.x >> 5;
  const int lane = threadIdx.x & 31;
  const int nloc = lane & 15;
  const int ncol = blockIdx.y * 64 + wv * 16 + nloc;
  const int half = (lane >> 4) & 1;
  const int kbA  = half * 8;
  const int moff = half * 8;
  int mrow = m0 + nloc; if (mrow >= n) mrow = n - 1;

  v8f acc = {};
#pragma unroll
  for (int s = 0; s < 8; ++s) {
    const float* ap = h0 + (size_t)mrow * 256 + s * 32 + kbA;
    const float4 f0 = *(const float4*)(ap);
    const float4 f1 = *(const float4*)(ap + 4);
    const float4 f2 = *(const float4*)(ap + 16);
    const float4 f3 = *(const float4*)(ap + 20);
    v16bf a;
    a[0] = (bf16_t)f0.x; a[1] = (bf16_t)f0.y; a[2]  = (bf16_t)f0.z; a[3]  = (bf16_t)f0.w;
    a[4] = (bf16_t)f1.x; a[5] = (bf16_t)f1.y; a[6]  = (bf16_t)f1.z; a[7]  = (bf16_t)f1.w;
    a[8] = (bf16_t)f2.x; a[9] = (bf16_t)f2.y; a[10] = (bf16_t)f2.z; a[11] = (bf16_t)f2.w;
    a[12] = (bf16_t)f3.x; a[13] = (bf16_t)f3.y; a[14] = (bf16_t)f3.z; a[15] = (bf16_t)f3.w;
    const v16bf b = *(const v16bf*)(w + ((size_t)(s * 256 + ncol) * 2 + half) * 16);
    acc = __builtin_amdgcn_wmma_f32_16x16x32_bf16(false, a, false, b, (short)0, acc, false, false);
  }
  const float bb = b1[ncol], g = gamma[ncol], be = beta[ncol];
  const float mu = mean[ncol], iv = rsqrtf(var[ncol] + 1e-5f);
#pragma unroll
  for (int r = 0; r < 8; ++r) {
    int m = m0 + moff + r;
    if (m < n) {
      float c = fmaxf(acc[r] + bb, 0.0f);
      z1[(size_t)m * 256 + ncol] = (c - mu) * iv * g + be;
    }
  }
}

// logits: 256 -> 10, VALU dot products with W in LDS
__global__ __launch_bounds__(256) void logit_kernel(
    const float* __restrict__ z1, const float* __restrict__ w2,
    const float* __restrict__ b2, float* __restrict__ out, int n) {
  __shared__ float ws[256 * 10];
  for (int i = threadIdx.x; i < 2560; i += 256) ws[i] = w2[i];
  __syncthreads();
  int nid = blockIdx.x * 256 + threadIdx.x;
  if (nid >= n) return;
  float acc[10] = {};
  const float* zr = z1 + (size_t)nid * 256;
  for (int k = 0; k < 256; ++k) {
    float zv = zr[k];
#pragma unroll
    for (int c = 0; c < 10; ++c) acc[c] += zv * ws[k * 10 + c];
  }
#pragma unroll
  for (int c = 0; c < 10; ++c) out[(size_t)nid * 10 + c] = acc[c] + b2[c];
}

// ---------------------------------------------------------------- launch
extern "C" void kernel_launch(void* const* d_in, const int* in_sizes, int n_in,
                              void* d_out, int out_size, void* d_ws, size_t ws_size,
                              hipStream_t stream) {
  const float* features = (const float*)d_in[0];
  const int*   eidx     = (const int*)d_in[1];
  const float* edgenet  = (const float*)d_in[2];
  const float* pae_w1   = (const float*)d_in[7];
  const float* pae_b1   = (const float*)d_in[8];
  const float* pae_w2   = (const float*)d_in[9];
  const float* pae_b2   = (const float*)d_in[10];
  const float* cheb_w[4] = {(const float*)d_in[11], (const float*)d_in[12],
                            (const float*)d_in[13], (const float*)d_in[14]};
  const float* cls_w1 = (const float*)d_in[15];
  const float* cls_b1 = (const float*)d_in[16];
  const float* bn_g   = (const float*)d_in[17];
  const float* bn_b   = (const float*)d_in[18];
  const float* bn_m   = (const float*)d_in[19];
  const float* bn_v   = (const float*)d_in[20];
  const float* cls_w2 = (const float*)d_in[21];
  const float* cls_b2 = (const float*)d_in[22];
  float* out = (float*)d_out;

  const int N = in_sizes[0] / D_IN;     // 100000
  const int E = in_sizes[1] / 2;        // 1600000
  const int* rowp = eidx;
  const int* colp = eidx + E;

  // workspace carve-up (256B aligned)
  char* wsb = (char*)d_ws;
  size_t off = 0;
  auto carve = [&](size_t bytes) -> char* {
    char* p = wsb + off;
    off = (off + bytes + 255) & ~(size_t)255;
    return p;
  };
  bf16_t* pw1  = (bf16_t*)carve(16384 * 2);
  bf16_t* pw2  = (bf16_t*)carve(16384 * 2);
  bf16_t* cw[4];
  cw[0] = (bf16_t*)carve(3 * 128 * 64 * 2);
  cw[1] = (bf16_t*)carve(3 * 64 * 64 * 2);
  cw[2] = (bf16_t*)carve(3 * 64 * 64 * 2);
  cw[3] = (bf16_t*)carve(3 * 64 * 64 * 2);
  bf16_t* cls1 = (bf16_t*)carve(256 * 256 * 2);
  float* ew   = (float*)carve((size_t)E * 4);
  float* nrm  = (float*)carve((size_t)E * 4);
  float* deg  = (float*)carve((size_t)N * 4);
  float* dinv = (float*)carve((size_t)N * 4);
  float* tbuf = (float*)carve((size_t)N * 256 * 4);  // t1 | t2, aliased as z1
  float* t1 = tbuf;
  float* t2 = tbuf + (size_t)N * 128;
  float* z1 = tbuf;

  // --- weight conversions to swizzled bf16 (fragment-major)
  cvt_swz_kernel<<<(16384 + 255) / 256, 256, 0, stream>>>(pae_w1, pw1, 128, 128);
  cvt_swz_kernel<<<(16384 + 255) / 256, 256, 0, stream>>>(pae_w2, pw2, 128, 128);
  cvt_swz_kernel<<<(24576 + 255) / 256, 256, 0, stream>>>(cheb_w[0], cw[0], 384, 64);
  cvt_swz_kernel<<<(12288 + 255) / 256, 256, 0, stream>>>(cheb_w[1], cw[1], 192, 64);
  cvt_swz_kernel<<<(12288 + 255) / 256, 256, 0, stream>>>(cheb_w[2], cw[2], 192, 64);
  cvt_swz_kernel<<<(12288 + 255) / 256, 256, 0, stream>>>(cheb_w[3], cw[3], 192, 64);
  cvt_swz_kernel<<<(65536 + 255) / 256, 256, 0, stream>>>(cls_w1, cls1, 256, 256);

  // --- PAE edge weights (WMMA + TDM double buffering)
  pae_kernel<<<(E + 63) / 64, 256, 0, stream>>>(edgenet, pw1, pae_b1, pw2, pae_b2, ew, E);

  // --- normalization
  hipMemsetAsync(deg, 0, (size_t)N * 4, stream);
  deg_kernel<<<(E + 255) / 256, 256, 0, stream>>>(rowp, ew, deg, E);
  dinv_kernel<<<(N + 255) / 256, 256, 0, stream>>>(deg, dinv, N);
  normc_kernel<<<(E + 255) / 256, 256, 0, stream>>>(rowp, colp, ew, dinv, nrm, E);

  // --- 4 Cheb layers
  const float* z = features;
  int ldz = D_IN;
  int din = D_IN;
  for (int i = 0; i < 4; ++i) {
    size_t zb = (size_t)N * din * 4;
    int chunks = din >> 2;
    hipMemsetAsync(t1, 0, zb, stream);
    spmm_kernel<<<((size_t)E * chunks + 255) / 256, 256, 0, stream>>>(
        rowp, colp, nrm, z, ldz, din, t1, E);
    hipMemsetAsync(t2, 0, zb, stream);
    spmm_kernel<<<((size_t)E * chunks + 255) / 256, 256, 0, stream>>>(
        rowp, colp, nrm, t1, din, din, t2, E);
    axpy_kernel<<<((size_t)N * din + 255) / 256, 256, 0, stream>>>(t2, z, ldz, din, N);
    cheb_gemm_kernel<<<(N + 15) / 16, 128, 0, stream>>>(
        z, ldz, t1, t2, din, cw[i], out, HGC * i, N);
    z = out + HGC * i;
    ldz = 256;
    din = HGC;
  }

  // --- classifier
  dim3 cgrid((N + 15) / 16, 4);
  cls_gemm_kernel<<<cgrid, 128, 0, stream>>>(out, cls1, cls_b1, bn_g, bn_b, bn_m, bn_v, z1, N);
  logit_kernel<<<(N + 255) / 256, 256, 0, stream>>>(z1, cls_w2, cls_b2,
                                                    out + (size_t)N * 256, N);
}
